// LowRankBilinearFusion_35648228556885
// MI455X (gfx1250) — matrix operand order repaired and checked by reading, not compiled, not run on hardware
//
#include <hip/hip_runtime.h>
#include <hip/hip_bf16.h>

typedef __attribute__((ext_vector_type(16))) _Float16 v16h;
typedef __attribute__((ext_vector_type(8)))  _Float16 v8h;
typedef __attribute__((ext_vector_type(8)))  float    v8f;

// ---------------------------------------------------------------------------
// Stage 0: convert Wp (fp32 256x256) to f16
// ---------------------------------------------------------------------------
__global__ void cvt_f32_to_f16_kernel(const float* __restrict__ in,
                                      _Float16* __restrict__ out, int n) {
    int i = blockIdx.x * blockDim.x + threadIdx.x;
    if (i < n) out[i] = (_Float16)in[i];
}

// ---------------------------------------------------------------------------
// Stage 1: C_f16[M,N] = A_f32[M,K] * B_f32[N,K]^T   (both K-contiguous)
// One wave per 16x16 output tile, K-loop in steps of 32 via wmma f16->f32.
// ---------------------------------------------------------------------------
__global__ void gemm_abT_f32f16_kernel(const float* __restrict__ A,
                                       const float* __restrict__ B,
                                       _Float16* __restrict__ C,
                                       int Mrows, int N, int K) {
    const int wave = threadIdx.x >> 5;
    const int lane = threadIdx.x & 31;
    const int tile = blockIdx.x * 8 + wave;
    const int ntn  = N >> 4;                 // tiles along N
    const int mt   = tile / ntn;
    const int nt   = tile % ntn;
    if (mt * 16 >= Mrows) return;

    const int r  = lane & 15;                // row (A) / col (B) within tile
    const int kh = (lane >> 4) * 8;          // A half-lane K shift
    const int kg = (lane >> 4) * 16;         // unused; B also uses kh

    const float* arow = A + (size_t)(mt * 16 + r) * K;
    const float* brow = B + (size_t)(nt * 16 + r) * K;

    v8f c = {};
    for (int k0 = 0; k0 < K; k0 += 32) {
        v16h af, bf;
#pragma unroll
        for (int i = 0; i < 8; ++i) {
            af[i]     = (_Float16)arow[k0 + kh + i];
            af[8 + i] = (_Float16)arow[k0 + kh + 16 + i];
        }
#pragma unroll
        for (int i = 0; i < 8; ++i) {
            bf[i]     = (_Float16)brow[k0 + kh + i];
            bf[8 + i] = (_Float16)brow[k0 + kh + 16 + i];
        }
        c = __builtin_amdgcn_wmma_f32_16x16x32_f16(false, af, false, bf,
                                                   (short)0, c, false, false);
    }

    const int mg = (lane >> 4) * 8;
    _Float16* crow = C + (size_t)(mt * 16 + mg) * N + nt * 16 + (lane & 15);
#pragma unroll
    for (int i = 0; i < 8; ++i) crow[(size_t)i * N] = (_Float16)c[i];
}

// ---------------------------------------------------------------------------
// Stage 2: fused bilinear + projection + bias + relu.
// Block = one (b,m) pair, 8 waves; wave w owns n-tile w (rows n = 16w..16w+15).
// A[n,k] = vp[b,n,k] * up[b,m,k]  (built in-register, 8 K-step fragments)
// O[n,f] = A * Wp^T; Wp staged through LDS in 64-row chunks (bank-padded).
// ---------------------------------------------------------------------------
#define SWP_STRIDE 264  // 256 + 8 halves pad -> 16B row shift, conflict-free

__global__ void fused_bilinear_kernel(const _Float16* __restrict__ up,
                                      const _Float16* __restrict__ vp,
                                      const _Float16* __restrict__ wp,
                                      const float* __restrict__ bp,
                                      float* __restrict__ out) {
    __shared__ _Float16 sWp[64 * SWP_STRIDE];   // 33 KB

    const int bm   = blockIdx.x;     // b*32 + m
    const int b    = bm >> 5;
    const int tid  = threadIdx.x;
    const int wave = tid >> 5;
    const int lane = tid & 31;
    const int nbase = wave * 16;

    // ---- build the 8 A fragments (K = 256 total, 32 per step) ----
    const int r  = lane & 15;
    const int kh = (lane >> 4) * 8;
    const _Float16* uprow = up + (size_t)bm * 256;
    const _Float16* vprow = vp + (size_t)(b * 128 + nbase + r) * 256;

    v16h afrag[8];
#pragma unroll
    for (int ks = 0; ks < 8; ++ks) {
        const int k0 = ks * 32 + kh;
        v8h u0 = *(const v8h*)(uprow + k0);
        v8h u1 = *(const v8h*)(uprow + k0 + 16);
        v8h x0 = *(const v8h*)(vprow + k0);
        v8h x1 = *(const v8h*)(vprow + k0 + 16);
        v8h p0 = u0 * x0;
        v8h p1 = u1 * x1;
        afrag[ks] = __builtin_shufflevector(p0, p1, 0, 1, 2, 3, 4, 5, 6, 7,
                                            8, 9, 10, 11, 12, 13, 14, 15);
    }

    const int ncol = lane & 15;          // f within tile / N index
    const int mg   = (lane >> 4) * 8;    // n row-group within tile
    const int kg   = (lane >> 4) * 8;    // B half-lane K shift
    float* orow = out + ((size_t)bm * 128 + nbase) * 256;

    // ---- 4 chunks of 64 Wp rows (4 f-tiles each) ----
    for (int chunk = 0; chunk < 4; ++chunk) {
        __syncthreads();
        // cooperative stage: 64 rows x 256 halves, 32B per work item
        for (int i = tid; i < 64 * 16; i += 256) {
            const int fr  = i >> 4;
            const int seg = i & 15;
            const v8h* src = (const v8h*)(wp + (size_t)(chunk * 64 + fr) * 256 + seg * 16);
            v8h* dst = (v8h*)(sWp + fr * SWP_STRIDE + seg * 16);
            dst[0] = src[0];
            dst[1] = src[1];
        }
        __syncthreads();

#pragma unroll
        for (int ftl = 0; ftl < 4; ++ftl) {
            const int ft = chunk * 4 + ftl;
            v8f c = {};
#pragma unroll
            for (int ks = 0; ks < 8; ++ks) {
                const _Float16* pb = sWp + (ftl * 16 + ncol) * SWP_STRIDE + ks * 32 + kg;
                v8h blo = *(const v8h*)pb;
                v8h bhi = *(const v8h*)(pb + 16);
                v16h bfrag = __builtin_shufflevector(blo, bhi, 0, 1, 2, 3, 4, 5, 6, 7,
                                                     8, 9, 10, 11, 12, 13, 14, 15);
                c = __builtin_amdgcn_wmma_f32_16x16x32_f16(false, afrag[ks], false, bfrag,
                                                           (short)0, c, false, false);
            }
            const float bv = bp[ft * 16 + ncol];
#pragma unroll
            for (int i = 0; i < 8; ++i) {
                float val = c[i] + bv;
                orow[(size_t)(mg + i) * 256 + ft * 16 + ncol] = val > 0.0f ? val : 0.0f;
            }
        }
    }
}

// ---------------------------------------------------------------------------
// Launch
// ---------------------------------------------------------------------------
extern "C" void kernel_launch(void* const* d_in, const int* in_sizes, int n_in,
                              void* d_out, int out_size, void* d_ws, size_t ws_size,
                              hipStream_t stream) {
    const float* u  = (const float*)d_in[0];   // (32,32,1024)
    const float* v  = (const float*)d_in[1];   // (32,128,2048)
    const float* Wu = (const float*)d_in[2];   // (256,1024)
    const float* Wv = (const float*)d_in[3];   // (256,2048)
    const float* Wp = (const float*)d_in[4];   // (256,256)
    const float* bp = (const float*)d_in[5];   // (256,)
    float* out = (float*)d_out;                // (32,32,128,256)

    char* ws = (char*)d_ws;
    _Float16* up_h = (_Float16*)(ws);                       // 1024*256*2 = 512 KB
    _Float16* vp_h = (_Float16*)(ws + 524288);              // 4096*256*2 = 2 MB
    _Float16* wp_h = (_Float16*)(ws + 524288 + 2097152);    // 256*256*2 = 128 KB

    // Stage 0: Wp -> f16
    cvt_f32_to_f16_kernel<<<256, 256, 0, stream>>>(Wp, wp_h, 256 * 256);

    // Stage 1: projections (WMMA, f32 inputs converted on the fly)
    // up: (1024 x 256) = u(1024x1024) x Wu^T; 1024 tiles / 8 waves = 128 blocks
    gemm_abT_f32f16_kernel<<<128, 256, 0, stream>>>(u, Wu, up_h, 1024, 256, 1024);
    // vp: (4096 x 256) = v(4096x2048) x Wv^T; 4096 tiles / 8 waves = 512 blocks
    gemm_abT_f32f16_kernel<<<512, 256, 0, stream>>>(v, Wv, vp_h, 4096, 256, 2048);

    // Stage 2: fused bilinear product + p-projection + bias + relu
    fused_bilinear_kernel<<<1024, 256, 0, stream>>>(up_h, vp_h, wp_h, bp, out);
}
